// RNN_72997264163157
// MI455X (gfx1250) — matrix operand order repaired and checked by reading, not compiled, not run on hardware
//
#include <hip/hip_runtime.h>
#include <hip/hip_bf16.h>
#include <math.h>

// ---------------- problem constants ----------------
#define BATCH   64
#define TFULL   128
#define TSTEPS  127
#define VOCAB   10000
#define VPAD    10240
#define EDIM    300
#define EPAD    320
#define HDIM    1024
#define GDIM    4096   // 4*H

typedef __bf16 bf16;
typedef __attribute__((ext_vector_type(16))) __bf16 v16bf;
typedef __attribute__((ext_vector_type(8)))  __bf16 v8bf;
typedef __attribute__((ext_vector_type(8)))  float  v8f;

// ---------------- CDNA5 async global->LDS copy (ASYNCcnt tracked) ----------------
// Per-lane: copy 16 bytes from global to LDS. lds_off is the 32-bit LDS byte
// address (low 32 bits of the generic shared-aperture address).
__device__ __forceinline__ void async_copy_b128(unsigned int lds_off, const void* gptr) {
  asm volatile("global_load_async_to_lds_b128 %0, %1, off"
               :: "v"(lds_off), "v"(gptr) : "memory");
}
__device__ __forceinline__ void wait_async0() {
  asm volatile("s_wait_asynccnt 0x0" ::: "memory");
}

// ---------------- WMMA tile loads ----------------
// A: 16x32 bf16 tile (M x K) from an LDS slab (16 rows, leading dim lda).
// Per ISA layout: lanes 0-15 hold M=lane, K chunks [0:8) and [16:24);
// lanes 16-31 hold M=lane-16, K chunks [8:16) and [24:32).
__device__ __forceinline__ v16bf load_a16_lds(const bf16* __restrict__ sA, int lda,
                                              int k0, int lane) {
  const int m  = lane & 15;
  const int kh = (lane >> 4) * 8;
  const bf16* p = sA + m * lda + k0 + kh;
  v8bf lo = *(const v8bf*)(p);
  v8bf hi = *(const v8bf*)(p + 16);
  v16bf a;
#pragma unroll
  for (int i = 0; i < 8; ++i) { a[i] = lo[i]; a[i + 8] = hi[i]; }
  return a;
}

// B: 32x16 bf16 tile (K x N) from transposed weights Wt[n][k] (ld = ldw).
// Lane (n = lane&15, h = lane>>4) loads K chunks [8h:8h+8) and [16+8h:24+8h).
__device__ __forceinline__ v16bf load_b16(const bf16* __restrict__ Wt, int ldw,
                                          int n0, int k0, int lane) {
  const int n  = n0 + (lane & 15);
  const int kk = k0 + (lane >> 4) * 8;
  const bf16* p = Wt + (size_t)n * ldw + kk;
  v8bf lo = *(const v8bf*)(p);
  v8bf hi = *(const v8bf*)(p + 16);
  v16bf b;
#pragma unroll
  for (int i = 0; i < 8; ++i) { b[i] = lo[i]; b[i + 8] = hi[i]; }
  return b;
}

// ---------------- weight cast / transpose (one-time per launch) ----------------
// W[k][n] fp32 -> Wt[n][k] bf16, zero padded to (Npad, Kpad).
__global__ __launch_bounds__(256)
void cast_transpose_kernel(const float* __restrict__ W, bf16* __restrict__ Wt,
                           int K, int N, int Kpad, int Npad) {
  size_t total = (size_t)Kpad * Npad;
  for (size_t idx = (size_t)blockIdx.x * blockDim.x + threadIdx.x; idx < total;
       idx += (size_t)gridDim.x * blockDim.x) {
    int k = (int)(idx / Npad);
    int n = (int)(idx % Npad);
    float v = (k < K && n < N) ? W[(size_t)k * N + n] : 0.0f;
    Wt[(size_t)n * Kpad + k] = (bf16)v;
  }
}

// ---------------- embedding gather: X[t][b][k] bf16, k padded to EPAD ----------------
__global__ __launch_bounds__(256)
void embed_gather_kernel(const int* __restrict__ sent,
                         const float* __restrict__ wordvec,
                         bf16* __restrict__ X) {
  size_t total = (size_t)TSTEPS * BATCH * EPAD;
  for (size_t idx = (size_t)blockIdx.x * blockDim.x + threadIdx.x; idx < total;
       idx += (size_t)gridDim.x * blockDim.x) {
    int t = (int)(idx / (BATCH * EPAD));
    int r = (int)(idx % (BATCH * EPAD));
    int b = r / EPAD;
    int k = r % EPAD;
    int tok = sent[b * TFULL + t];
    X[idx] = (k < EDIM) ? (bf16)wordvec[(size_t)tok * EDIM + k] : (bf16)0.0f;
  }
}

// ---------------- zero init of recurrent state + loss accumulators ----------------
__global__ __launch_bounds__(256)
void init_state_kernel(bf16* __restrict__ h0, bf16* __restrict__ h1,
                       float* __restrict__ c0, float* __restrict__ c1,
                       float* __restrict__ sent_acc) {
  int idx = blockIdx.x * blockDim.x + threadIdx.x;
  if (idx < BATCH * HDIM) {
    h0[idx] = (bf16)0.0f;
    h1[idx] = (bf16)0.0f;
    c0[idx] = 0.0f;
    c1[idx] = 0.0f;
  }
  if (idx < BATCH) sent_acc[idx] = 0.0f;
}

// ---------------- gates GEMM: out(64x4096) = A1@W1t + A2@W2t + bias ----------------
// WG = one 16-row M slab x 128 columns, 8 waves, one 16x16 tile per wave.
// grid.x = (GDIM/128) * (BATCH/16) = 32*4 = 128.
// A slabs are staged into LDS with async global->LDS copies; K-loop vmem is
// then pure weight streaming (L2-resident bf16).
__global__ __launch_bounds__(256)
void gemm_gates_kernel(const bf16* __restrict__ A1, int lda1,  // 64 x lda1 (Kpad)
                       const bf16* __restrict__ W1t,           // [4096][lda1]
                       const bf16* __restrict__ A2,            // 64 x 1024
                       const bf16* __restrict__ W2t,           // [4096][1024]
                       const float* __restrict__ bias,
                       float* __restrict__ out) {
  extern __shared__ char smem_raw[];
  const int lane = threadIdx.x & 31;
  const int wave = threadIdx.x >> 5;
  const int mrow = blockIdx.x & 3;       // M slab: rows mrow*16 .. +15
  const int nblk = blockIdx.x >> 2;
  const int n0   = nblk * 128 + wave * 16;

  bf16* sA1 = (bf16*)smem_raw;                                   // 16 x lda1
  bf16* sA2 = (bf16*)(smem_raw + (size_t)16 * lda1 * sizeof(bf16)); // 16 x HDIM

  // ---- stage A slabs into LDS (async, ASYNCcnt) ----
  {
    const char* g1 = (const char*)(A1 + (size_t)mrow * 16 * lda1);
    const char* g2 = (const char*)(A2 + (size_t)mrow * 16 * HDIM);
    unsigned int l1 = (unsigned int)(uintptr_t)sA1;
    unsigned int l2 = (unsigned int)(uintptr_t)sA2;
    const int bytes1 = 16 * lda1 * (int)sizeof(bf16);
    const int bytes2 = 16 * HDIM * (int)sizeof(bf16);
    for (int ofs = threadIdx.x * 16; ofs < bytes1; ofs += 256 * 16)
      async_copy_b128(l1 + ofs, g1 + ofs);
    for (int ofs = threadIdx.x * 16; ofs < bytes2; ofs += 256 * 16)
      async_copy_b128(l2 + ofs, g2 + ofs);
    wait_async0();
    __syncthreads();
  }

  v8f acc = {};
  // x @ w_ih  (K = lda1 = 320 or 1024, zero padded so extra K contributes 0)
  for (int k0 = 0; k0 < lda1; k0 += 32) {
    __builtin_prefetch(W1t + (size_t)(n0 + (lane & 15)) * lda1 + k0 + 32, 0, 3);
    v16bf a = load_a16_lds(sA1, lda1, k0, lane);
    v16bf b = load_b16(W1t, lda1, n0, k0, lane);
    acc = __builtin_amdgcn_wmma_f32_16x16x32_bf16(false, a, false, b,
                                                  (short)0, acc, false, false);
  }
  // h @ w_hh  (K = 1024)
  for (int k0 = 0; k0 < HDIM; k0 += 32) {
    __builtin_prefetch(W2t + (size_t)(n0 + (lane & 15)) * HDIM + k0 + 32, 0, 3);
    v16bf a = load_a16_lds(sA2, HDIM, k0, lane);
    v16bf b = load_b16(W2t, HDIM, n0, k0, lane);
    acc = __builtin_amdgcn_wmma_f32_16x16x32_bf16(false, a, false, b,
                                                  (short)0, acc, false, false);
  }

  const int n = n0 + (lane & 15);
  const int m = mrow * 16 + (lane >> 4) * 8;
  const float bv = bias[n];
#pragma unroll
  for (int r = 0; r < 8; ++r)
    out[(size_t)(m + r) * GDIM + n] = acc[r] + bv;
}

// ---------------- LSTM elementwise update ----------------
__global__ __launch_bounds__(256)
void lstm_update_kernel(const float* __restrict__ gates,
                        float* __restrict__ c, bf16* __restrict__ h) {
  int idx = blockIdx.x * blockDim.x + threadIdx.x;  // 0..65535
  int b = idx >> 10;
  int n = idx & (HDIM - 1);
  const float* g = gates + (size_t)b * GDIM;
  float gi = g[n];
  float gf = g[n + 1024];
  float gg = g[n + 2048];
  float go = g[n + 3072];
  float si = 1.0f / (1.0f + __expf(-gi));
  float sf = 1.0f / (1.0f + __expf(-gf));
  float so = 1.0f / (1.0f + __expf(-go));
  float c2 = sf * c[idx] + si * tanhf(gg);
  c[idx] = c2;
  h[idx] = (bf16)(so * tanhf(c2));
}

// ---------------- logits GEMM: temp[b][t][:] = h1 @ w_out + b_out ----------------
// WG = 16-row M slab x 128 columns; grid.x = (VPAD/128)*4 = 320.
__global__ __launch_bounds__(256)
void gemm_logits_kernel(const bf16* __restrict__ H,       // 64x1024
                        const bf16* __restrict__ Wt,      // [10240][1024]
                        const float* __restrict__ bias,   // [10000]
                        float* __restrict__ outbase, int t) {
  extern __shared__ char smem_raw[];
  const int lane = threadIdx.x & 31;
  const int wave = threadIdx.x >> 5;
  const int mrow = blockIdx.x & 3;
  const int nblk = blockIdx.x >> 2;
  const int n0   = nblk * 128 + wave * 16;

  bf16* sA = (bf16*)smem_raw;  // 16 x HDIM

  {
    const char* g = (const char*)(H + (size_t)mrow * 16 * HDIM);
    unsigned int l = (unsigned int)(uintptr_t)sA;
    const int bytes = 16 * HDIM * (int)sizeof(bf16);
    for (int ofs = threadIdx.x * 16; ofs < bytes; ofs += 256 * 16)
      async_copy_b128(l + ofs, g + ofs);
    wait_async0();
    __syncthreads();
  }

  v8f acc = {};
  for (int k0 = 0; k0 < HDIM; k0 += 32) {
    __builtin_prefetch(Wt + (size_t)(n0 + (lane & 15)) * HDIM + k0 + 32, 0, 3);
    v16bf a = load_a16_lds(sA, HDIM, k0, lane);
    v16bf b = load_b16(Wt, HDIM, n0, k0, lane);
    acc = __builtin_amdgcn_wmma_f32_16x16x32_bf16(false, a, false, b,
                                                  (short)0, acc, false, false);
  }

  const int n = n0 + (lane & 15);
  const int m = mrow * 16 + (lane >> 4) * 8;
  if (n < VOCAB) {
    const float bv = bias[n];
#pragma unroll
    for (int r = 0; r < 8; ++r) {
      int b = m + r;  // batch index
      outbase[((size_t)b * TSTEPS + t) * VOCAB + n] = acc[r] + bv;
    }
  }
}

// ---------------- per-(b,t) online logsumexp + NLL accumulation ----------------
__global__ __launch_bounds__(256)
void loss_kernel(const float* __restrict__ logits_base,  // [B][T-1][V]
                 const int* __restrict__ sent,
                 float* __restrict__ sent_acc) {
  const int row = blockIdx.x;        // b*127 + t
  const int b = row / TSTEPS;
  const int t = row % TSTEPS;
  const float* lg = logits_base + (size_t)row * VOCAB;

  float m = -INFINITY, s = 0.0f;
  for (int v = threadIdx.x; v < VOCAB; v += 256) {
    float x = lg[v];
    float nm = fmaxf(m, x);
    s = s * __expf(m - nm) + __expf(x - nm);
    m = nm;
  }
  __shared__ float sm[256];
  __shared__ float ss[256];
  sm[threadIdx.x] = m;
  ss[threadIdx.x] = s;
  __syncthreads();
  for (int off = 128; off > 0; off >>= 1) {
    if (threadIdx.x < off) {
      float m2 = sm[threadIdx.x + off], s2 = ss[threadIdx.x + off];
      float nm = fmaxf(sm[threadIdx.x], m2);
      ss[threadIdx.x] = ss[threadIdx.x] * __expf(sm[threadIdx.x] - nm) +
                        s2 * __expf(m2 - nm);
      sm[threadIdx.x] = nm;
    }
    __syncthreads();
  }
  if (threadIdx.x == 0) {
    int gt = sent[b * TFULL + (t + 1)];
    if (gt != 0) {  // PAD_ID == 0
      float lse = sm[0] + __logf(ss[0]);
      atomicAdd(&sent_acc[b], -(lg[gt] - lse));
    }
  }
}

__global__ void final_loss_kernel(const float* __restrict__ sent_acc,
                                  const int* __restrict__ length,
                                  float* __restrict__ out0) {
  __shared__ float red[BATCH];
  int b = threadIdx.x;
  red[b] = sent_acc[b] / (float)length[b];
  __syncthreads();
  for (int off = 32; off > 0; off >>= 1) {
    if (b < off) red[b] += red[b + off];
    __syncthreads();
  }
  if (b == 0) out0[0] = red[0];
}

// ---------------- host orchestration ----------------
static inline size_t align256(size_t x) { return (x + 255) & ~(size_t)255; }

extern "C" void kernel_launch(void* const* d_in, const int* in_sizes, int n_in,
                              void* d_out, int out_size, void* d_ws, size_t ws_size,
                              hipStream_t stream) {
  (void)in_sizes; (void)n_in; (void)out_size; (void)ws_size;
  const int*   sent    = (const int*)d_in[0];
  const int*   length  = (const int*)d_in[1];
  const float* wordvec = (const float*)d_in[2];
  const float* w_ih0   = (const float*)d_in[3];
  const float* w_hh0   = (const float*)d_in[4];
  const float* b0      = (const float*)d_in[5];
  const float* w_ih1   = (const float*)d_in[6];
  const float* w_hh1   = (const float*)d_in[7];
  const float* b1      = (const float*)d_in[8];
  const float* w_out   = (const float*)d_in[9];
  const float* b_out   = (const float*)d_in[10];

  float* outf = (float*)d_out;
  float* logits_base = outf + 1;

  // workspace carve-up
  char* ws = (char*)d_ws;
  size_t off = 0;
  bf16* WihT0 = (bf16*)(ws + off); off = align256(off + (size_t)GDIM * EPAD * 2);
  bf16* WhhT0 = (bf16*)(ws + off); off = align256(off + (size_t)GDIM * HDIM * 2);
  bf16* WihT1 = (bf16*)(ws + off); off = align256(off + (size_t)GDIM * HDIM * 2);
  bf16* WhhT1 = (bf16*)(ws + off); off = align256(off + (size_t)GDIM * HDIM * 2);
  bf16* WoutT = (bf16*)(ws + off); off = align256(off + (size_t)VPAD * HDIM * 2);
  bf16* X     = (bf16*)(ws + off); off = align256(off + (size_t)TSTEPS * BATCH * EPAD * 2);
  float* gates0 = (float*)(ws + off); off = align256(off + (size_t)BATCH * GDIM * 4);
  float* gates1 = (float*)(ws + off); off = align256(off + (size_t)BATCH * GDIM * 4);
  bf16* h0 = (bf16*)(ws + off); off = align256(off + (size_t)BATCH * HDIM * 2);
  bf16* h1 = (bf16*)(ws + off); off = align256(off + (size_t)BATCH * HDIM * 2);
  float* c0 = (float*)(ws + off); off = align256(off + (size_t)BATCH * HDIM * 4);
  float* c1 = (float*)(ws + off); off = align256(off + (size_t)BATCH * HDIM * 4);
  float* sent_acc = (float*)(ws + off); off = align256(off + (size_t)BATCH * 4);

  // one-time (per launch) weight cast+transpose into bf16, K-contiguous
  cast_transpose_kernel<<<2048, 256, 0, stream>>>(w_ih0, WihT0, EDIM, GDIM, EPAD, GDIM);
  cast_transpose_kernel<<<4096, 256, 0, stream>>>(w_hh0, WhhT0, HDIM, GDIM, HDIM, GDIM);
  cast_transpose_kernel<<<4096, 256, 0, stream>>>(w_ih1, WihT1, HDIM, GDIM, HDIM, GDIM);
  cast_transpose_kernel<<<4096, 256, 0, stream>>>(w_hh1, WhhT1, HDIM, GDIM, HDIM, GDIM);
  cast_transpose_kernel<<<8192, 256, 0, stream>>>(w_out, WoutT, HDIM, VOCAB, HDIM, VPAD);

  embed_gather_kernel<<<4096, 256, 0, stream>>>(sent, wordvec, X);
  init_state_kernel<<<(BATCH * HDIM + 255) / 256, 256, 0, stream>>>(h0, h1, c0, c1, sent_acc);

  const size_t smem_g0 = (size_t)(16 * EPAD + 16 * HDIM) * sizeof(bf16);  // 43 KB
  const size_t smem_g1 = (size_t)(16 * HDIM + 16 * HDIM) * sizeof(bf16);  // 64 KB
  const size_t smem_lg = (size_t)(16 * HDIM) * sizeof(bf16);              // 32 KB

  for (int t = 0; t < TSTEPS; ++t) {
    const bf16* xt = X + (size_t)t * BATCH * EPAD;
    // layer 0
    gemm_gates_kernel<<<128, 256, smem_g0, stream>>>(xt, EPAD, WihT0,
                                                     h0, WhhT0, b0, gates0);
    lstm_update_kernel<<<(BATCH * HDIM) / 256, 256, 0, stream>>>(gates0, c0, h0);
    // layer 1 (input = new h0)
    gemm_gates_kernel<<<128, 256, smem_g1, stream>>>(h0, HDIM, WihT1,
                                                     h1, WhhT1, b1, gates1);
    lstm_update_kernel<<<(BATCH * HDIM) / 256, 256, 0, stream>>>(gates1, c1, h1);
    // logits for this timestep -> d_out temp
    gemm_logits_kernel<<<320, 256, smem_lg, stream>>>(h1, WoutT, b_out,
                                                      logits_base, t);
  }

  loss_kernel<<<BATCH * TSTEPS, 256, 0, stream>>>(logits_base, sent, sent_acc);
  final_loss_kernel<<<1, BATCH, 0, stream>>>(sent_acc, length, outf);
}
